// MoEBlock_55697135895244
// MI455X (gfx1250) — matrix operand order repaired and checked by reading, not compiled, not run on hardware
//
#include <hip/hip_runtime.h>
#include <hip/hip_bf16.h>

// ---------------------------------------------------------------------------
// Transformer block (LN -> causal MHA -> residual -> LN -> top2 MoE -> residual)
// for MI455X / gfx1250. All GEMMs run on v_wmma_f32_16x16x32_bf16 (fp32 accum).
// GEMM: 128x128x32 block tiles, 8 waves (2x4), 64x32 per wave (8 WMMA/K-step),
// double-buffered LDS with register staging for global/WMMA overlap.
// ---------------------------------------------------------------------------

#define BB   2
#define TT   2048
#define NN   1024
#define HH   16
#define HD   64
#define EE   8
#define FFD  4096
#define BT   (BB*TT)        // 4096 tokens

typedef __attribute__((ext_vector_type(16))) __bf16 bf16x16;
typedef __attribute__((ext_vector_type(8)))  __bf16 bf16x8;
typedef __attribute__((ext_vector_type(8)))  float  v8f;

// ---------------------------------------------------------------- helpers ---

__device__ __forceinline__ v8f wmma_bf16(bf16x16 a, bf16x16 b, v8f c) {
  // D = A(16x32 bf16) * B(32x16 bf16) + C(16x16 f32)
  return __builtin_amdgcn_wmma_f32_16x16x32_bf16(false, a, false, b,
                                                 (short)0, c, false, false);
}

// A-fragment per ISA 7.12.2: lane holds row (lane&15); elems 0..7 -> K = 8*(lane>>4)+j,
// elems 8..15 -> K = 16 + 8*(lane>>4) + j.  Two 16B contiguous chunks.
__device__ __forceinline__ bf16x16 ldA16(const __bf16* lo, const __bf16* hi) {
  bf16x8 l = *(const bf16x8*)lo;
  bf16x8 h = *(const bf16x8*)hi;
  bf16x16 r;
#pragma unroll
  for (int i = 0; i < 8; ++i) { r[i] = l[i]; r[8 + i] = h[i]; }
  return r;
}

// B-fragment: lane holds col (lane&15); elems 0..7 -> K = 8*(lane>>4)+j, elems 8..15 -> K = 16 + 8*(lane>>4) + j.
__device__ __forceinline__ bf16x16 ldB16(const __bf16* p) {
  bf16x8 l = *(const bf16x8*)p;
  bf16x8 h = *(const bf16x8*)(p + 16);
  bf16x16 r;
#pragma unroll
  for (int i = 0; i < 8; ++i) { r[i] = l[i]; r[8 + i] = h[i]; }
  return r;
}

__device__ __forceinline__ float gelu_tanh(float x) {
  const float c = 0.7978845608028654f;
  float t = tanhf(c * (x + 0.044715f * x * x * x));
  return 0.5f * x * (1.0f + t);
}

__device__ __forceinline__ float wave_sum32(float v) {
#pragma unroll
  for (int o = 16; o > 0; o >>= 1) v += __shfl_xor(v, o, 32);
  return v;
}

// ----------------------------------------------- weight fp32 -> bf16 [N][K] --

// in: [R][C] fp32 row-major (per blockIdx.z matrix); out: [C][R] bf16
__global__ __launch_bounds__(256) void transpose_cvt_kernel(
    const float* __restrict__ in, __bf16* __restrict__ out, int R, int C) {
  __shared__ float tile[32][33];
  const float* inp = in  + (size_t)blockIdx.z * R * C;
  __bf16*     outp = out + (size_t)blockIdx.z * R * C;
  int c0 = blockIdx.x * 32, r0 = blockIdx.y * 32;
#pragma unroll
  for (int i = threadIdx.y; i < 32; i += 8)
    tile[i][threadIdx.x] = inp[(size_t)(r0 + i) * C + c0 + threadIdx.x];
  __syncthreads();
#pragma unroll
  for (int i = threadIdx.y; i < 32; i += 8)
    outp[(size_t)(c0 + i) * R + r0 + threadIdx.x] = (__bf16)tile[threadIdx.x][i];
}

// ------------------------------------------------------------- layernorm ----

__global__ __launch_bounds__(256) void ln_bf16_kernel(
    const float* __restrict__ x, const float* __restrict__ g,
    const float* __restrict__ bta, __bf16* __restrict__ out) {
  __shared__ float s1[8], s2[8], sm[2];
  int row = blockIdx.x;
  const float4* xr = (const float4*)(x + (size_t)row * NN);
  float4 v = xr[threadIdx.x];
  float a = v.x + v.y + v.z + v.w;
  float b = v.x * v.x + v.y * v.y + v.z * v.z + v.w * v.w;
  a = wave_sum32(a); b = wave_sum32(b);
  int wid = threadIdx.x >> 5, ln = threadIdx.x & 31;
  if (ln == 0) { s1[wid] = a; s2[wid] = b; }
  __syncthreads();
  if (threadIdx.x == 0) {
    float ta = 0.f, tb = 0.f;
#pragma unroll
    for (int i = 0; i < 8; ++i) { ta += s1[i]; tb += s2[i]; }
    float mu  = ta * (1.0f / NN);
    float var = tb * (1.0f / NN) - mu * mu;
    sm[0] = mu; sm[1] = rsqrtf(var + 1e-5f);
  }
  __syncthreads();
  float mu = sm[0], inv = sm[1];
  int base = threadIdx.x * 4;
  float xv[4] = {v.x, v.y, v.z, v.w};
#pragma unroll
  for (int j = 0; j < 4; ++j)
    out[(size_t)row * NN + base + j] =
        (__bf16)((xv[j] - mu) * inv * g[base + j] + bta[base + j]);
}

// -------------------------------------------------------- WMMA bf16 GEMM ----
// C[128x128 per block] = A[M x K](bf16, optionally row-gathered) * Bt^T + bias.
// Bt is the weight stored TRANSPOSED: bf16 [N][K] so LDS fill and B-fragment
// reads are contiguous. 8 waves arranged 2(M) x 4(N); each wave owns 64x32 of C
// = 4x2 accumulators -> 8 WMMA per 32-deep K step. Double-buffered LDS with
// register staging so global loads for step k+1 overlap WMMA for step k.

#define EPI_BIAS_F32  0
#define EPI_RESID     1
#define EPI_GELU_BF16 2
#define EPI_SCATTER   3

template <int EPI>
__global__ __launch_bounds__(256) void gemm_bf16_kernel(
    const __bf16* __restrict__ A, const __bf16* __restrict__ Bt,
    const float* __restrict__ bias,
    float* __restrict__ outF, float* __restrict__ outF2,
    __bf16* __restrict__ outB, const float* __restrict__ resid,
    const int* __restrict__ gather_idx, const int* __restrict__ cntPtr,
    const int* __restrict__ scat_idx, const float* __restrict__ scat_p,
    int Mmax, int Nn, int Kk, int ldc) {
  __shared__ __align__(16) __bf16 As[2][128][40];
  __shared__ __align__(16) __bf16 Bs[2][128][40];

  int Mc = Mmax;
  if (cntPtr) { int c = *cntPtr; Mc = c < Mmax ? c : Mmax; }
  int m0 = blockIdx.y * 128;
  int n0 = blockIdx.x * 128;
  if (m0 >= Mc) return;   // uniform early-exit for empty expert tiles

  int lane = threadIdx.x & 31, wid = threadIdx.x >> 5;
  int wm = wid >> 2, wn = wid & 3;              // 2 x 4 wave grid
  int lr = lane & 15, lh = lane >> 4;

  v8f acc[4][2];
#pragma unroll
  for (int mi = 0; mi < 4; ++mi)
#pragma unroll
    for (int ni = 0; ni < 2; ++ni)
#pragma unroll
      for (int i = 0; i < 8; ++i) acc[mi][ni][i] = 0.f;

  // Staging map: 512 16B-chunks per tile; thread covers chunks t and t+256.
  int r0c = threadIdx.x >> 2;                   // 0..63
  int r1c = r0c + 64;                           // 64..127
  int seg = threadIdx.x & 3;                    // 0..3 (k offset seg*8)

  // Gather rows are K-invariant: resolve once.
  int grow0 = -1, grow1 = -1;
  if (m0 + r0c < Mc) grow0 = gather_idx ? gather_idx[m0 + r0c] : (m0 + r0c);
  if (m0 + r1c < Mc) grow1 = gather_idx ? gather_idx[m0 + r1c] : (m0 + r1c);
  const __bf16* ap0 = (grow0 >= 0) ? (A + (size_t)grow0 * Kk + seg * 8) : nullptr;
  const __bf16* ap1 = (grow1 >= 0) ? (A + (size_t)grow1 * Kk + seg * 8) : nullptr;
  const __bf16* bp0 = Bt + (size_t)(n0 + r0c) * Kk + seg * 8;
  const __bf16* bp1 = Bt + (size_t)(n0 + r1c) * Kk + seg * 8;

  const uint4 z4 = make_uint4(0u, 0u, 0u, 0u);
  uint4 ra0 = ap0 ? *(const uint4*)ap0 : z4;
  uint4 ra1 = ap1 ? *(const uint4*)ap1 : z4;
  uint4 rb0 = *(const uint4*)bp0;
  uint4 rb1 = *(const uint4*)bp1;
  *(uint4*)&As[0][r0c][seg * 8] = ra0;
  *(uint4*)&As[0][r1c][seg * 8] = ra1;
  *(uint4*)&Bs[0][r0c][seg * 8] = rb0;
  *(uint4*)&Bs[0][r1c][seg * 8] = rb1;
  __syncthreads();

  int p = 0;
  for (int k0 = 0; k0 < Kk; k0 += 32) {
    bool has_next = (k0 + 32 < Kk);
    if (has_next) {                              // issue global loads early
      ra0 = ap0 ? *(const uint4*)(ap0 + k0 + 32) : z4;
      ra1 = ap1 ? *(const uint4*)(ap1 + k0 + 32) : z4;
      rb0 = *(const uint4*)(bp0 + k0 + 32);
      rb1 = *(const uint4*)(bp1 + k0 + 32);
      if (k0 + 64 < Kk) __builtin_prefetch(bp0 + k0 + 64, 0, 1);  // global_prefetch_b8
    }
    // ---- 8 WMMA from LDS stage p ----
    bf16x16 af[4], bfr[2];
#pragma unroll
    for (int mi = 0; mi < 4; ++mi) {
      int r = wm * 64 + mi * 16 + lr;
      af[mi] = ldA16(&As[p][r][lh * 8], &As[p][r][16 + lh * 8]);
    }
#pragma unroll
    for (int ni = 0; ni < 2; ++ni) {
      int c = wn * 32 + ni * 16 + lr;
      bfr[ni] = ldB16(&Bs[p][c][lh * 8]);
    }
#pragma unroll
    for (int mi = 0; mi < 4; ++mi)
#pragma unroll
      for (int ni = 0; ni < 2; ++ni)
        acc[mi][ni] = wmma_bf16(af[mi], bfr[ni], acc[mi][ni]);

    if (has_next) {                              // fill the other stage
      *(uint4*)&As[p ^ 1][r0c][seg * 8] = ra0;
      *(uint4*)&As[p ^ 1][r1c][seg * 8] = ra1;
      *(uint4*)&Bs[p ^ 1][r0c][seg * 8] = rb0;
      *(uint4*)&Bs[p ^ 1][r1c][seg * 8] = rb1;
      __syncthreads();
      p ^= 1;
    }
  }

  // ---- epilogue: C layout col = (lane&15), row = v + 8*(lane>>4)
#pragma unroll
  for (int mi = 0; mi < 4; ++mi) {
    int mb = m0 + wm * 64 + mi * 16 + 8 * lh;
#pragma unroll
    for (int ni = 0; ni < 2; ++ni) {
      int n = n0 + wn * 32 + ni * 16 + lr;
      float bi = bias[n];
#pragma unroll
      for (int v = 0; v < 8; ++v) {
        int m = mb + v;
        if (m >= Mc) continue;
        float val = acc[mi][ni][v] + bi;
        if (EPI == EPI_BIAS_F32) {
          outF[(size_t)m * ldc + n] = val;
        } else if (EPI == EPI_RESID) {
          float r = resid[(size_t)m * ldc + n] + val;   // SCALE = 1.0
          outF [(size_t)m * ldc + n] = r;
          outF2[(size_t)m * ldc + n] = r;
        } else if (EPI == EPI_GELU_BF16) {
          outB[(size_t)m * ldc + n] = (__bf16)gelu_tanh(val);
        } else { // EPI_SCATTER: d_out[token] += p*(acc+bias); token unique/launch
          int   tok = scat_idx[m];
          float pw  = scat_p[m];
          outF[(size_t)tok * ldc + n] += pw * val;
        }
      }
    }
  }
}

// ------------------------------------------------- QKV split / cast kernel --

__global__ __launch_bounds__(256) void split_qkv_kernel(
    const float* __restrict__ qkv, __bf16* __restrict__ Qb,
    __bf16* __restrict__ Kb, __bf16* __restrict__ Vt) {
  size_t i = (size_t)blockIdx.x * 256 + threadIdx.x;
  if (i >= (size_t)BT * NN) return;
  int d = (int)(i & 63);
  int h = (int)((i >> 6) & 15);
  size_t tok = i >> 10;
  int b = (int)(tok >> 11);
  int t = (int)(tok & 2047);
  const float* row = qkv + tok * (3 * NN);
  size_t bh = (size_t)b * HH + h;
  Qb[(bh * TT + t) * HD + d] = (__bf16)row[h * HD + d];
  Kb[(bh * TT + t) * HD + d] = (__bf16)row[NN + h * HD + d];
  Vt[(bh * HD + d) * TT + t] = (__bf16)row[2 * NN + h * HD + d];   // V transposed [d][t]
}

// ------------------------------------------------ flash attention (wave32) --
// One wave per (b, h, 16-query tile). QK^T and PV both on WMMA; online softmax
// with 16-lane shfl row-reductions; P repacked to A-fragment layout via LDS.

__global__ __launch_bounds__(32) void attn_fa_kernel(
    const __bf16* __restrict__ Qb, const __bf16* __restrict__ Kb,
    const __bf16* __restrict__ Vt, __bf16* __restrict__ Yb) {
  __shared__ __align__(16) __bf16 Pl[16][40];
  int lane = threadIdx.x;
  int q0 = blockIdx.x * 16;
  int h = blockIdx.y, b = blockIdx.z;
  size_t bh = (size_t)b * HH + h;
  const __bf16* Qh = Qb + bh * TT * HD;
  const __bf16* Kh = Kb + bh * TT * HD;
  const __bf16* Vh = Vt + bh * HD * TT;

  int lr = lane & 15, lh = lane >> 4;
  int qrow = q0 + lr;
  int akb = lh * 8;
  bf16x16 qa0 = ldA16(Qh + (size_t)qrow * HD + akb,      Qh + (size_t)qrow * HD + 16 + akb);
  bf16x16 qa1 = ldA16(Qh + (size_t)qrow * HD + 32 + akb, Qh + (size_t)qrow * HD + 48 + akb);

  float mrun[8], lrun[8], al[8];
  v8f o[4];
#pragma unroll
  for (int v = 0; v < 8; ++v) { mrun[v] = -1e30f; lrun[v] = 0.f; }
#pragma unroll
  for (int t = 0; t < 4; ++t)
#pragma unroll
    for (int v = 0; v < 8; ++v) o[t][v] = 0.f;

  const float sc = 0.125f;               // 1/sqrt(64)
  int ktmax = (q0 + 15) >> 5;
  for (int kt = 0; kt <= ktmax; ++kt) {
    int key0 = kt * 32;
    v8f s0, s1;
#pragma unroll
    for (int v = 0; v < 8; ++v) { s0[v] = 0.f; s1[v] = 0.f; }
    {
      int bkb = lh * 8;                  // contraction runs over d
      const __bf16* k0p = Kh + (size_t)(key0 + lr) * HD + bkb;
      s0 = wmma_bf16(qa0, ldB16(k0p), s0);
      s0 = wmma_bf16(qa1, ldB16(k0p + 32), s0);
      const __bf16* k1p = Kh + (size_t)(key0 + 16 + lr) * HD + bkb;
      s1 = wmma_bf16(qa0, ldB16(k1p), s1);
      s1 = wmma_bf16(qa1, ldB16(k1p + 32), s1);
    }
    // online softmax across 32 keys
    int c0 = key0 + lr, c1 = c0 + 16;
#pragma unroll
    for (int v = 0; v < 8; ++v) {
      int q = q0 + v + 8 * lh;
      float x0 = (c0 <= q) ? s0[v] * sc : -1e30f;
      float x1 = (c1 <= q) ? s1[v] * sc : -1e30f;
      float mx = fmaxf(x0, x1);
#pragma unroll
      for (int off = 8; off > 0; off >>= 1) mx = fmaxf(mx, __shfl_xor(mx, off, 16));
      float mn = fmaxf(mrun[v], mx);
      float p0 = __expf(x0 - mn), p1 = __expf(x1 - mn);
      float rs = p0 + p1;
#pragma unroll
      for (int off = 8; off > 0; off >>= 1) rs += __shfl_xor(rs, off, 16);
      al[v]   = __expf(mrun[v] - mn);
      lrun[v] = lrun[v] * al[v] + rs;
      mrun[v] = mn;
      Pl[v + 8 * lh][lr]      = (__bf16)p0;   // repack P into A-frag layout
      Pl[v + 8 * lh][16 + lr] = (__bf16)p1;
    }
#pragma unroll
    for (int t = 0; t < 4; ++t)
#pragma unroll
      for (int v = 0; v < 8; ++v) o[t][v] *= al[v];
    __syncthreads();                     // single-wave: orders LDS via dscnt wait
    bf16x16 pf = ldA16(&Pl[lr][akb], &Pl[lr][16 + akb]);
#pragma unroll
    for (int t = 0; t < 4; ++t) {
      int d = t * 16 + lr;
      bf16x16 vb = ldB16(Vh + (size_t)d * TT + key0 + lh * 8);
      o[t] = wmma_bf16(pf, vb, o[t]);
    }
    __syncthreads();
  }
#pragma unroll
  for (int t = 0; t < 4; ++t)
#pragma unroll
    for (int v = 0; v < 8; ++v) {
      int q = q0 + v + 8 * lh;
      float val = o[t][v] / lrun[v];
      Yb[((size_t)(b * TT + q)) * NN + h * HD + t * 16 + lr] = (__bf16)val;
    }
}

// ------------------------------------------------------ gating + routing ----

__global__ __launch_bounds__(32) void gate_kernel(
    const __bf16* __restrict__ xb, const float* __restrict__ wg,
    const float* __restrict__ bg, int* __restrict__ gi, float* __restrict__ gp) {
  int tok = blockIdx.x, l = threadIdx.x;
  float acc[EE];
#pragma unroll
  for (int e = 0; e < EE; ++e) acc[e] = 0.f;
  for (int d = l; d < NN; d += 32) {
    float xv = (float)xb[(size_t)tok * NN + d];
#pragma unroll
    for (int e = 0; e < EE; ++e) acc[e] += xv * wg[d * EE + e];
  }
#pragma unroll
  for (int e = 0; e < EE; ++e) acc[e] = wave_sum32(acc[e]);
  if (l == 0) {
    float pr[EE], mx = -1e30f, se = 0.f;
#pragma unroll
    for (int e = 0; e < EE; ++e) { acc[e] += bg[e]; mx = fmaxf(mx, acc[e]); }
#pragma unroll
    for (int e = 0; e < EE; ++e) { pr[e] = __expf(acc[e] - mx); se += pr[e]; }
    int i0 = 0;
#pragma unroll
    for (int e = 1; e < EE; ++e) if (pr[e] > pr[i0]) i0 = e;
    int i1 = (i0 == 0) ? 1 : 0;
#pragma unroll
    for (int e = 0; e < EE; ++e) if (e != i0 && pr[e] > pr[i1]) i1 = e;
    gi[tok * 2]     = i0;  gp[tok * 2]     = pr[i0] / se;
    gi[tok * 2 + 1] = i1;  gp[tok * 2 + 1] = pr[i1] / se;
  }
}

__global__ void zero_counts_kernel(int* counts) {
  if (threadIdx.x < EE) counts[threadIdx.x] = 0;
}

__global__ __launch_bounds__(256) void route_kernel(
    const int* __restrict__ gi, const float* __restrict__ gp,
    int* __restrict__ counts, int* __restrict__ rows, float* __restrict__ probs) {
  int i = blockIdx.x * 256 + threadIdx.x;
  if (i >= BT * 2) return;
  int e = gi[i];
  int pos = atomicAdd(&counts[e], 1);
  rows [e * BT + pos] = i >> 1;
  probs[e * BT + pos] = gp[i];
}

// ------------------------------------------------------------------ launch --

extern "C" void kernel_launch(void* const* d_in, const int* in_sizes, int n_in,
                              void* d_out, int out_size, void* d_ws, size_t ws_size,
                              hipStream_t stream) {
  (void)in_sizes; (void)n_in; (void)out_size; (void)ws_size;
  const float* x      = (const float*)d_in[0];
  const float* ln1_s  = (const float*)d_in[1];
  const float* ln1_b  = (const float*)d_in[2];
  const float* w_qkv  = (const float*)d_in[3];
  const float* b_qkv  = (const float*)d_in[4];
  const float* w_ap   = (const float*)d_in[5];
  const float* b_ap   = (const float*)d_in[6];
  const float* ln2_s  = (const float*)d_in[7];
  const float* ln2_b  = (const float*)d_in[8];
  const float* w_gate = (const float*)d_in[9];
  const float* b_gate = (const float*)d_in[10];
  const float* w_fc   = (const float*)d_in[11];
  const float* b_fc   = (const float*)d_in[12];
  const float* w_fp   = (const float*)d_in[13];
  const float* b_fp   = (const float*)d_in[14];
  float* out = (float*)d_out;

  // ---- workspace bump allocator (≈295 MB) ----
  char* base = (char*)d_ws;
  size_t off = 0;
  auto alloc = [&](size_t bytes) -> char* {
    char* p = base + off;
    off = (off + bytes + 255) & ~(size_t)255;
    return p;
  };
  __bf16* wqkv_t = (__bf16*)alloc((size_t)NN * 3 * NN * 2);       // [3N][N]
  __bf16* wap_t  = (__bf16*)alloc((size_t)NN * NN * 2);           // [N][N]
  __bf16* wfc_t  = (__bf16*)alloc((size_t)EE * FFD * NN * 2);     // [E][FF][N]
  __bf16* wfp_t  = (__bf16*)alloc((size_t)EE * NN * FFD * 2);     // [E][N][FF]
  __bf16* xa     = (__bf16*)alloc((size_t)BT * NN * 2);
  float*  qkvf   = (float*) alloc((size_t)BT * 3 * NN * 4);
  __bf16* Qb     = (__bf16*)alloc((size_t)BT * NN * 2);
  __bf16* Kb     = (__bf16*)alloc((size_t)BT * NN * 2);
  __bf16* Vt     = (__bf16*)alloc((size_t)BT * NN * 2);
  __bf16* attnb  = (__bf16*)alloc((size_t)BT * NN * 2);
  float*  x1f    = (float*) alloc((size_t)BT * NN * 4);
  __bf16* xb     = (__bf16*)alloc((size_t)BT * NN * 2);
  int*    gi     = (int*)   alloc((size_t)BT * 2 * 4);
  float*  gp     = (float*) alloc((size_t)BT * 2 * 4);
  int*    counts = (int*)   alloc((size_t)EE * 4);
  int*    rows   = (int*)   alloc((size_t)EE * BT * 4);
  float*  probs  = (float*) alloc((size_t)EE * BT * 4);
  __bf16* hbuf   = (__bf16*)alloc((size_t)BT * FFD * 2);

  dim3 tb(32, 8);
  // 1) weights -> bf16, transposed to [N][K]
  transpose_cvt_kernel<<<dim3(3 * NN / 32, NN / 32, 1), tb, 0, stream>>>(w_qkv, wqkv_t, NN, 3 * NN);
  transpose_cvt_kernel<<<dim3(NN / 32, NN / 32, 1),     tb, 0, stream>>>(w_ap,  wap_t,  NN, NN);
  transpose_cvt_kernel<<<dim3(FFD / 32, NN / 32, EE),   tb, 0, stream>>>(w_fc,  wfc_t,  NN, FFD);
  transpose_cvt_kernel<<<dim3(NN / 32, FFD / 32, EE),   tb, 0, stream>>>(w_fp,  wfp_t,  FFD, NN);

  // 2) LN1 -> bf16
  ln_bf16_kernel<<<BT, 256, 0, stream>>>(x, ln1_s, ln1_b, xa);

  // 3) QKV GEMM: [4096x1024] x [1024x3072]
  gemm_bf16_kernel<EPI_BIAS_F32><<<dim3(3 * NN / 128, BT / 128), 256, 0, stream>>>(
      xa, wqkv_t, b_qkv, qkvf, nullptr, nullptr, nullptr,
      nullptr, nullptr, nullptr, nullptr, BT, 3 * NN, NN, 3 * NN);

  // 4) split/cast Q,K,V (V transposed for PV B-fragments)
  split_qkv_kernel<<<(BT * NN + 255) / 256, 256, 0, stream>>>(qkvf, Qb, Kb, Vt);

  // 5) flash attention, one wave per 16-query tile
  attn_fa_kernel<<<dim3(TT / 16, HH, BB), 32, 0, stream>>>(Qb, Kb, Vt, attnb);

  // 6) attn proj + residual: x1 = x + (attn @ Wp + bp); also seeds d_out
  gemm_bf16_kernel<EPI_RESID><<<dim3(NN / 128, BT / 128), 256, 0, stream>>>(
      attnb, wap_t, b_ap, x1f, out, nullptr, x,
      nullptr, nullptr, nullptr, nullptr, BT, NN, NN, NN);

  // 7) LN2 -> bf16
  ln_bf16_kernel<<<BT, 256, 0, stream>>>(x1f, ln2_s, ln2_b, xb);

  // 8) gating + routing (device-side compaction per expert)
  gate_kernel<<<BT, 32, 0, stream>>>(xb, w_gate, b_gate, gi, gp);
  zero_counts_kernel<<<1, 32, 0, stream>>>(counts);
  route_kernel<<<(BT * 2 + 255) / 256, 256, 0, stream>>>(gi, gp, counts, rows, probs);

  // 9) experts: gathered fc (+GELU) then proj with scatter-add into d_out
  for (int e = 0; e < EE; ++e) {
    gemm_bf16_kernel<EPI_GELU_BF16><<<dim3(FFD / 128, BT / 128), 256, 0, stream>>>(
        xb, wfc_t + (size_t)e * FFD * NN, b_fc + (size_t)e * FFD,
        nullptr, nullptr, hbuf, nullptr,
        rows + e * BT, counts + e, nullptr, nullptr, BT, FFD, NN, FFD);
    gemm_bf16_kernel<EPI_SCATTER><<<dim3(NN / 128, BT / 128), 256, 0, stream>>>(
        hbuf, wfp_t + (size_t)e * NN * FFD, b_fp + (size_t)e * NN,
        out, nullptr, nullptr, nullptr,
        nullptr, counts + e, rows + e * BT, probs + e * BT, BT, NN, FFD, NN);
  }
}
